// SpatialClassifierVN_42279658062115
// MI455X (gfx1250) — compile-verified, matrix operand order checked by reading, not run on hardware
//
#include <hip/hip_runtime.h>
#include <hip/hip_bf16.h>

// ---------------- problem constants (match reference) ----------------
#define NQC   8192          // NQ
#define NCC   8192          // NC
#define ETOT  (NQC * 32)    // E
#define ECCH  64            // EC
#define CUT   10.0f
#define NCLS_ 13

#if defined(__has_builtin)
#if __has_builtin(__builtin_amdgcn_tensor_load_to_lds)
#define HAVE_TDM 1
#else
#define HAVE_TDM 0
#endif
#else
#define HAVE_TDM 0
#endif

// ---------------- d_in flat index map ----------------
// top-level dict order, then params pytree (sorted dict keys, recursive):
enum {
  IN_POSQ = 0, IN_POSC, IN_NSCA, IN_NVEC, IN_EIDX,
  P_W_E2N, P_W_EVN, P_W_EXP, P_W_N2E, P_W_SCA, P_B_E2N, P_B_N2E, P_B_SCA,
  PC_GVL_WG, PC_GVL_WS, PC_GVL_WV1, PC_GVL_WV2, PC_GVL_BG,            // cls_gvl
  PCP_WDIR, PCP_WG, PCP_WS, PCP_WV1, PCP_WV2, PCP_BG,                  // cls_gvp
  PE_WDIR, PE_WG, PE_WS, PE_WV1, PE_WV2, PE_BG,                        // edge_gvp
  PN_WG, PN_WS, PN_WV1, PN_WV2, PN_BG,                                  // node_gvl
  PO_WG, PO_WS, PO_WV1, PO_WV2, PO_BG                                   // out_gvl
};

typedef __attribute__((ext_vector_type(16))) _Float16 v16h;
typedef __attribute__((ext_vector_type(8)))  float    v8f;
typedef __attribute__((ext_vector_type(4)))  unsigned u32x4;
typedef __attribute__((ext_vector_type(4)))  int      i32x4;
typedef __attribute__((ext_vector_type(8)))  int      i32x8;

struct P {
  // raw inputs
  const float* pos_q; const float* pos_c;
  const float* node_sca; const float* node_vec;
  const int* edge_index;
  // f32 params read directly in kernels
  const float *W_exp, *b_sca, *b_e2n, *b_n2e;
  const float *node_bg, *edge_bg, *out_bg, *cls_bg;
  const float *cls_gvl_Ws;      // (13,160) f32, final matvec
  // f16 converted weights (workspace)
  const _Float16 *hWv1_n, *hWv2_n, *hWs_n, *hWg_n;                 // node_gvl
  const _Float16 *hWdir_e, *hWg_e, *hWs_e, *hWv1_e, *hWv2_e;       // edge_gvp
  const _Float16 *hWsca, *hWe2n, *hWn2e, *hWevn;                   // message
  const _Float16 *hWg_o, *hWs_o, *hWv1_o, *hWv2_o;                 // out_gvl
  const _Float16 *hWdir_c, *hWg_c, *hWs_c, *hWv1_c, *hWv2_c;       // cls_gvp
  const _Float16 *hWv1_f;                                           // cls_gvl
  // intermediates
  _Float16 *nsall;   // (NC,128) f16
  _Float16 *nvall;   // (NC,3,32) f16  (row = 96 contiguous f16)
  float *agg_s;      // (NQ,128)
  float *agg_v;      // (NQ,3,32)
  float *out;        // (NQ,13)
};

#define FORI(i, n) for (int i = threadIdx.x; i < (n); i += blockDim.x)

// ---------------- WMMA tile GEMM helper ----------------
// Fragment loader: 16-bit A-matrix 16x32 layout (ISA 7.12.2):
//   lane = half*16 + idx; row = idx; VGPR r<4: K = half*8+2r(+1); r>=4: K = 16+half*8+2(r-4)(+1)
// Loaded as 8 dword LDS/global reads per fragment.
union FragU { v16h v; unsigned u[8]; };

__device__ __forceinline__ v16h load_frag(const _Float16* base, int ld, int row,
                                          int k0, int half_) {
  const unsigned* pu = (const unsigned*)(base + row * ld + k0);
  FragU f;
  const int o = half_ * 4;
#pragma unroll
  for (int r = 0; r < 4; ++r) { f.u[r] = pu[o + r]; f.u[r + 4] = pu[8 + o + r]; }
  return f.v;
}

// C[16][N] (f32, ldc) = A[16][K] (f16 row-major, lda) * B[N][K]^T (f16 (out,in) row-major, ldb)
// All waves of the block cooperate on n-tiles. Call from all threads; sync outside.
__device__ __forceinline__ void gemm16(const _Float16* A, int lda,
                                       const _Float16* B, int ldb,
                                       float* C, int ldc, int N, int K) {
  const int lane  = threadIdx.x & 31;
  const int wave  = threadIdx.x >> 5;
  const int nw    = blockDim.x >> 5;
  const int half_ = lane >> 4;
  const int idx   = lane & 15;
  for (int nt = wave * 16; nt < N; nt += nw * 16) {
    v8f acc = {};
    for (int k0 = 0; k0 < K; k0 += 32) {
      v16h a = load_frag(A, lda, idx, k0, half_);
      v16h b = load_frag(B, ldb, nt + idx, k0, half_);
      acc = __builtin_amdgcn_wmma_f32_16x16x32_f16(false, a, false, b,
                                                   (short)0, acc, false, false);
    }
#pragma unroll
    for (int r = 0; r < 8; ++r)
      C[(r + 8 * half_) * ldc + nt + idx] = acc[r];
  }
}

// ---------------- small utility kernels ----------------
__global__ void cvt_kernel(const float* __restrict__ s, _Float16* __restrict__ d, int n) {
  int i = blockIdx.x * blockDim.x + threadIdx.x;
  if (i < n) d[i] = (_Float16)s[i];
}
__global__ void zero_kernel(float* __restrict__ p, int n) {
  int i = blockIdx.x * blockDim.x + threadIdx.x;
  if (i < n) p[i] = 0.f;
}

// ---------------- node precompute: ns_all / nv_all ----------------
__global__ __launch_bounds__(128) void node_kernel(P p) {
  __shared__ alignas(16) _Float16 sV[3][16][64];
  __shared__ alignas(16) _Float16 sVH[3][16][64];
  __shared__ alignas(16) _Float16 sX[16][320];     // [vnorm | sca]
  __shared__ alignas(16) _Float16 sS16[16][128];
  __shared__ float sF[16][128];
  __shared__ float sN[16][64];
  __shared__ float sOV[3][16][32];
  __shared__ float sG[16][32];

  const int n0 = blockIdx.x * 16;

  FORI(i, 16 * 64) ((float*)sN)[i] = 0.f;
  FORI(i, 16 * 64 * 3) {
    int c = i % 64, m = (i / 64) % 16, k = i / 1024;
    sV[k][m][c] = (_Float16)p.node_vec[((size_t)(n0 + m) * 64 + c) * 3 + k];
  }
  FORI(i, 16 * 256) {
    int c = i % 256, m = i / 256;
    sX[m][64 + c] = (_Float16)p.node_sca[(size_t)(n0 + m) * 256 + c];
  }
  __syncthreads();

  // vh = W_v1 @ vec  (3 x 16x64x64)
  for (int k = 0; k < 3; ++k) {
    gemm16(&sV[k][0][0], 64, p.hWv1_n, 64, &sF[0][0], 128, 64, 64);
    __syncthreads();
    FORI(i, 16 * 64) {
      int c = i % 64, m = i / 64;
      float v = sF[m][c];
      sVH[k][m][c] = (_Float16)v;
      sN[m][c] += v * v;
    }
    __syncthreads();
  }
  FORI(i, 16 * 64) { int c = i % 64, m = i / 64; sX[m][c] = (_Float16)sqrtf(sN[m][c]); }
  __syncthreads();

  // out_s = [vnorm|sca] @ W_s^T  (16x128x320)
  gemm16(&sX[0][0], 320, p.hWs_n, 320, &sF[0][0], 128, 128, 320);
  __syncthreads();
  FORI(i, 16 * 128) {
    int c = i % 128, m = i / 128;
    _Float16 h = (_Float16)sF[m][c];
    sS16[m][c] = h;
    p.nsall[(size_t)(n0 + m) * 128 + c] = h;
  }
  __syncthreads();

  // gate logits + out_v
  gemm16(&sS16[0][0], 128, p.hWg_n, 128, &sG[0][0], 32, 32, 128);
  for (int k = 0; k < 3; ++k)
    gemm16(&sVH[k][0][0], 64, p.hWv2_n, 64, &sOV[k][0][0], 32, 32, 64);
  __syncthreads();
  FORI(i, 16 * 32) {
    int c = i % 32, m = i / 32;
    float g = 1.f / (1.f + __expf(-(sG[m][c] + p.node_bg[c])));
    for (int k = 0; k < 3; ++k)
      p.nvall[((size_t)(n0 + m) * 3 + k) * 32 + c] = (_Float16)(g * sOV[k][m][c]);
  }
}

// ---------------- fused per-edge pipeline + scatter ----------------
__global__ __launch_bounds__(128) void edge_kernel(P p) {
  __shared__ int   sQI[16], sCI[16];
  __shared__ float sDist[16], sCut[16], sU[16][3];
  __shared__ alignas(16) _Float16 hA[3][16][64];   // edge_vec -> gated v -> ev
  __shared__ alignas(16) _Float16 hB[3][16][64];   // vh_e -> d -> vh2 (cols 0..31)
  __shared__ alignas(16) _Float16 hXe[16][128];    // [vnorm|gauss], later h_s f16
  __shared__ alignas(16) _Float16 h64[16][64];     // out_s f16 -> es f16
  __shared__ alignas(16) _Float16 hNS[16][128];    // gathered ns rows (TDM target)
  __shared__ alignas(16) _Float16 hNV[16][3][32];  // gathered nv rows (TDM target)
  __shared__ alignas(16) _Float16 hYV[3][16][32];
  __shared__ alignas(16) _Float16 hX2[16][160];    // [vnorm2 | y_s]
  __shared__ float fS[16][128];
  __shared__ float fa[16][64];
  __shared__ float fb[16][64];
  __shared__ float fc[16][64];
  __shared__ float fg[16][32];
  __shared__ float fg2[16][32];
  __shared__ float sE2N[16][32], sN2E[16][32];
#define FB32(m, c) ((float*)fb)[(m) * 32 + (c)]

  const int e0 = blockIdx.x * 16;
  const float step  = CUT / (ECCH - 1);
  const float coeff = -0.5f / (step * step);

#if HAVE_TDM
  // ---- Tensor Data Mover: async gather of 16 node rows into LDS, issued
  // up-front so the DMA overlaps the whole edge-GVP GEMM chain.
  if (threadIdx.x < 32) {
    const int* eci = p.edge_index + ETOT + e0;
    int pk[8];
#pragma unroll
    for (int j = 0; j < 8; ++j) {
      int a = eci[2 * j] & 0xffff;
      int b = eci[2 * j + 1] & 0xffff;
      pk[j] = __builtin_amdgcn_readfirstlane(a | (b << 16));
    }
    i32x4 g2 = { pk[0], pk[1], pk[2], pk[3] };   // gather row indices 0..7
    i32x4 g3 = { pk[4], pk[5], pk[6], pk[7] };   // gather row indices 8..15
    i32x8 gz = { 0, 0, 0, 0, 0, 0, 0, 0 };       // unused trailing descriptor group
    // D# group0: count=1, gather_mode=1 (16-bit idx), lds_addr, global_addr, type=2
    {
      unsigned long long ga = (unsigned long long)(uintptr_t)p.nsall;
      unsigned lds = (unsigned)(uintptr_t)&hNS[0][0];
      u32x4 g0 = { 0x80000001u, lds, (unsigned)ga,
                   ((unsigned)(ga >> 32) & 0x01ffffffu) | 0x80000000u };
      // group1: data_size=2B; tensor_dim0=128, tensor_dim1=NC; tile_dim0=128,
      // tile_dim1=16 indices; tensor_dim0_stride=128
      i32x8 g1 = { 0x10000, 128 << 16, (int)(8192u << 16), 128 << 16,
                   16, 128, 0, 0 };
      __builtin_amdgcn_tensor_load_to_lds(g0, g1, g2, g3, gz, 0);
    }
    {
      unsigned long long gv = (unsigned long long)(uintptr_t)p.nvall;
      unsigned lds = (unsigned)(uintptr_t)&hNV[0][0][0];
      u32x4 g0 = { 0x80000001u, lds, (unsigned)gv,
                   ((unsigned)(gv >> 32) & 0x01ffffffu) | 0x80000000u };
      i32x8 g1 = { 0x10000, 96 << 16, (int)(8192u << 16), 96 << 16,
                   16, 96, 0, 0 };
      __builtin_amdgcn_tensor_load_to_lds(g0, g1, g2, g3, gz, 0);
    }
  }
#endif

  // geometry
  if (threadIdx.x < 16) {
    int m = threadIdx.x, e = e0 + m;
    int q = p.edge_index[e], c = p.edge_index[ETOT + e];
    sQI[m] = q; sCI[m] = c;
    float dx = p.pos_q[q * 3 + 0] - p.pos_c[c * 3 + 0];
    float dy = p.pos_q[q * 3 + 1] - p.pos_c[c * 3 + 1];
    float dz = p.pos_q[q * 3 + 2] - p.pos_c[c * 3 + 2];
    float d = sqrtf(dx * dx + dy * dy + dz * dz);
    sDist[m] = d;
    float inv = 1.f / (d + 1e-7f);
    sU[m][0] = dx * inv; sU[m][1] = dy * inv; sU[m][2] = dz * inv;
    float C = 0.5f * (cosf(d * 3.14159265358979323846f / CUT) + 1.f);
    sCut[m] = (d <= CUT) ? C : 0.f;
  }
  FORI(i, 16 * 64) ((float*)fb)[i] = 0.f;
  __syncthreads();

  // gaussians + rank-1 edge_vec
  FORI(i, 16 * 64) {
    int c = i % 64, m = i / 64;
    float dd = sDist[m] - c * step;
    hXe[m][64 + c] = (_Float16)__expf(coeff * dd * dd);
    float w = p.W_exp[c];
    hA[0][m][c] = (_Float16)(sU[m][0] * w);
    hA[1][m][c] = (_Float16)(sU[m][1] * w);
    hA[2][m][c] = (_Float16)(sU[m][2] * w);
  }
  __syncthreads();

  // edge gvl: vh + vnorm
  for (int k = 0; k < 3; ++k) {
    gemm16(&hA[k][0][0], 64, p.hWv1_e, 64, &fa[0][0], 64, 64, 64);
    __syncthreads();
    FORI(i, 16 * 64) {
      int c = i % 64, m = i / 64;
      float v = fa[m][c];
      hB[k][m][c] = (_Float16)v;
      fb[m][c] += v * v;
    }
    __syncthreads();
  }
  FORI(i, 16 * 64) { int c = i % 64, m = i / 64; hXe[m][c] = (_Float16)sqrtf(fb[m][c]); }
  __syncthreads();

  // out_s (pre-activation kept for gate)
  gemm16(&hXe[0][0], 128, p.hWs_e, 128, &fc[0][0], 64, 64, 128);
  __syncthreads();
  FORI(i, 16 * 64) { int c = i % 64, m = i / 64; h64[m][c] = (_Float16)fc[m][c]; }
  __syncthreads();

  // gate, out_v -> gated vector x in hA
  gemm16(&h64[0][0], 64, p.hWg_e, 64, &fa[0][0], 64, 64, 64);
  __syncthreads();
  FORI(i, 16 * 64) { int c = i % 64, m = i / 64;
    fa[m][c] = 1.f / (1.f + __expf(-(fa[m][c] + p.edge_bg[c]))); }
  __syncthreads();
  for (int k = 0; k < 3; ++k) {
    gemm16(&hB[k][0][0], 64, p.hWv2_e, 64, &fb[0][0], 64, 64, 64);
    __syncthreads();
    FORI(i, 16 * 64) { int c = i % 64, m = i / 64;
      hA[k][m][c] = (_Float16)(fa[m][c] * fb[m][c]); }
    __syncthreads();
  }

  // es = leaky(out_s, 0.01)
  FORI(i, 16 * 64) { int c = i % 64, m = i / 64;
    float s = fc[m][c]; h64[m][c] = (_Float16)(s > 0.f ? s : 0.01f * s); }
  FORI(i, 16 * 64) { ((float*)fa)[i] = 0.f; ((float*)fc)[i] = 0.f; }  // dot / dsq
  __syncthreads();

  // VN leaky-relu on x (slope 0.2): d = W_dir x
  for (int k = 0; k < 3; ++k) {
    gemm16(&hA[k][0][0], 64, p.hWdir_e, 64, &fb[0][0], 64, 64, 64);
    __syncthreads();
    FORI(i, 16 * 64) { int c = i % 64, m = i / 64;
      float d = fb[m][c], x = (float)hA[k][m][c];
      hB[k][m][c] = (_Float16)d; fa[m][c] += x * d; fc[m][c] += d * d; }
    __syncthreads();
  }
  FORI(i, 16 * 64) {
    int c = i % 64, m = i / 64;
    float dot = fa[m][c], dsq = fc[m][c];
    for (int k = 0; k < 3; ++k) {
      float x = (float)hA[k][m][c], d = (float)hB[k][m][c];
      float neg = x - (dot / (dsq + 1e-6f)) * d;
      hA[k][m][c] = (_Float16)(0.2f * x + 0.8f * ((dot >= 0.f) ? x : neg));  // ev
    }
  }
  __syncthreads();

  // node features: TDM gather already in flight -> just drain TENSORcnt.
#if HAVE_TDM
  if (threadIdx.x < 32) __builtin_amdgcn_s_wait_tensorcnt(0);
  __syncthreads();
#else
  FORI(i, 16 * 128) { int c = i % 128, m = i / 128;
    hNS[m][c] = p.nsall[(size_t)sCI[m] * 128 + c]; }
  FORI(i, 16 * 96) { int c = i % 32, k = (i / 32) % 3, m = i / 96;
    hNV[m][k][c] = p.nvall[((size_t)sCI[m] * 3 + k) * 32 + c]; }
  __syncthreads();
#endif

  // y_s = ns * (es @ W_sca^T + b)
  gemm16(&h64[0][0], 64, p.hWsca, 64, &fS[0][0], 128, 128, 64);
  __syncthreads();
  FORI(i, 16 * 128) { int c = i % 128, m = i / 128;
    hX2[m][32 + c] = (_Float16)((float)hNS[m][c] * (fS[m][c] + p.b_sca[c])); }
  __syncthreads();

  // y_v = (es@We2n + b) * nv + (ns@Wn2e + b) * (W_evn ev)
  gemm16(&h64[0][0], 64, p.hWe2n, 64, &sE2N[0][0], 32, 32, 64);
  gemm16(&hNS[0][0], 128, p.hWn2e, 128, &sN2E[0][0], 32, 32, 128);
  __syncthreads();
  for (int k = 0; k < 3; ++k) {
    gemm16(&hA[k][0][0], 64, p.hWevn, 64, &fg[0][0], 32, 32, 64);
    __syncthreads();
    FORI(i, 16 * 32) { int c = i % 32, m = i / 32;
      float yv = (sE2N[m][c] + p.b_e2n[c]) * (float)hNV[m][k][c]
               + (sN2E[m][c] + p.b_n2e[c]) * fg[m][c];
      hYV[k][m][c] = (_Float16)yv; }
    __syncthreads();
  }

  // out_gvl
  FORI(i, 16 * 32) FB32(0, i) = 0.f;
  __syncthreads();
  for (int k = 0; k < 3; ++k) {
    gemm16(&hYV[k][0][0], 32, p.hWv1_o, 32, &fg[0][0], 32, 32, 32);
    __syncthreads();
    FORI(i, 16 * 32) { int c = i % 32, m = i / 32;
      float v = fg[m][c]; hB[k][m][c] = (_Float16)v; FB32(m, c) += v * v; }
    __syncthreads();
  }
  FORI(i, 16 * 32) { int c = i % 32, m = i / 32; hX2[m][c] = (_Float16)sqrtf(FB32(m, c)); }
  __syncthreads();
  gemm16(&hX2[0][0], 160, p.hWs_o, 160, &fS[0][0], 128, 128, 160);   // h_s
  __syncthreads();
  FORI(i, 16 * 128) { int c = i % 128, m = i / 128; hXe[m][c] = (_Float16)fS[m][c]; }
  __syncthreads();
  gemm16(&hXe[0][0], 128, p.hWg_o, 128, &fg2[0][0], 32, 32, 128);
  __syncthreads();
  FORI(i, 16 * 32) { int c = i % 32, m = i / 32;
    fg2[m][c] = 1.f / (1.f + __expf(-(fg2[m][c] + p.out_bg[c]))); }
  __syncthreads();

  // cutoff + scatter-add
  FORI(i, 16 * 128) { int c = i % 128, m = i / 128;
    atomicAdd(&p.agg_s[(size_t)sQI[m] * 128 + c], fS[m][c] * sCut[m]); }
  for (int k = 0; k < 3; ++k) {
    gemm16(&hB[k][0][0], 64, p.hWv2_o, 32, &fg[0][0], 32, 32, 32);
    __syncthreads();
    FORI(i, 16 * 32) { int c = i % 32, m = i / 32;
      atomicAdd(&p.agg_v[((size_t)sQI[m] * 3 + k) * 32 + c],
                fg2[m][c] * fg[m][c] * sCut[m]); }
    __syncthreads();
  }
#undef FB32
}

// ---------------- classifier ----------------
__global__ __launch_bounds__(128) void cls_kernel(P p) {
  __shared__ alignas(16) _Float16 aS[16][128];     // agg_s f16 -> leaky(s)
  __shared__ alignas(16) _Float16 aV[3][16][32];   // agg_v -> gated v -> ev
  __shared__ alignas(16) _Float16 hV[3][16][32];   // vh
  __shared__ alignas(16) _Float16 hD[3][16][32];   // d
  __shared__ alignas(16) _Float16 hX2[16][160];
  __shared__ alignas(16) _Float16 hHS[16][128];
  __shared__ float fS[16][128];
  __shared__ float fg[16][32], fg2[16][32], fn[16][32], fq[16][32];

  const int q0 = blockIdx.x * 16;
  FORI(i, 16 * 128) { int c = i % 128, m = i / 128;
    aS[m][c] = (_Float16)p.agg_s[(size_t)(q0 + m) * 128 + c]; }
  FORI(i, 16 * 96) { int c = i % 32, k = (i / 32) % 3, m = i / 96;
    aV[k][m][c] = (_Float16)p.agg_v[((size_t)(q0 + m) * 3 + k) * 32 + c]; }
  FORI(i, 16 * 32) ((float*)fn)[i] = 0.f;
  __syncthreads();

  // cls_gvp.gvl
  for (int k = 0; k < 3; ++k) {
    gemm16(&aV[k][0][0], 32, p.hWv1_c, 32, &fg[0][0], 32, 32, 32);
    __syncthreads();
    FORI(i, 16 * 32) { int c = i % 32, m = i / 32;
      float v = fg[m][c]; hV[k][m][c] = (_Float16)v; fn[m][c] += v * v; }
    __syncthreads();
  }
  FORI(i, 16 * 32) { int c = i % 32, m = i / 32; hX2[m][c] = (_Float16)sqrtf(fn[m][c]); }
  FORI(i, 16 * 128) { int c = i % 128, m = i / 128; hX2[m][32 + c] = aS[m][c]; }
  __syncthreads();
  gemm16(&hX2[0][0], 160, p.hWs_c, 160, &fS[0][0], 128, 128, 160);
  __syncthreads();
  FORI(i, 16 * 128) { int c = i % 128, m = i / 128; hHS[m][c] = (_Float16)fS[m][c]; }
  __syncthreads();
  gemm16(&hHS[0][0], 128, p.hWg_c, 128, &fg2[0][0], 32, 32, 128);
  __syncthreads();
  FORI(i, 16 * 32) { int c = i % 32, m = i / 32;
    fg2[m][c] = 1.f / (1.f + __expf(-(fg2[m][c] + p.cls_bg[c]))); }
  __syncthreads();
  for (int k = 0; k < 3; ++k) {
    gemm16(&hV[k][0][0], 32, p.hWv2_c, 32, &fg[0][0], 32, 32, 32);
    __syncthreads();
    FORI(i, 16 * 32) { int c = i % 32, m = i / 32;
      aV[k][m][c] = (_Float16)(fg2[m][c] * fg[m][c]); }  // gated v
    __syncthreads();
  }
  FORI(i, 16 * 128) { int c = i % 128, m = i / 128;
    float s = fS[m][c]; aS[m][c] = (_Float16)(s > 0.f ? s : 0.01f * s); }
  FORI(i, 16 * 32) { ((float*)fn)[i] = 0.f; ((float*)fq)[i] = 0.f; }
  __syncthreads();

  // VN leaky with W_dir_c
  for (int k = 0; k < 3; ++k) {
    gemm16(&aV[k][0][0], 32, p.hWdir_c, 32, &fg[0][0], 32, 32, 32);
    __syncthreads();
    FORI(i, 16 * 32) { int c = i % 32, m = i / 32;
      float d = fg[m][c], x = (float)aV[k][m][c];
      hD[k][m][c] = (_Float16)d; fn[m][c] += x * d; fq[m][c] += d * d; }
    __syncthreads();
  }
  FORI(i, 16 * 32) {
    int c = i % 32, m = i / 32;
    float dot = fn[m][c], dsq = fq[m][c];
    for (int k = 0; k < 3; ++k) {
      float x = (float)aV[k][m][c], d = (float)hD[k][m][c];
      float neg = x - (dot / (dsq + 1e-6f)) * d;
      aV[k][m][c] = (_Float16)(0.2f * x + 0.8f * ((dot >= 0.f) ? x : neg));
    }
  }
  FORI(i, 16 * 32) ((float*)fn)[i] = 0.f;
  __syncthreads();

  // cls_gvl: only vnorm(W_v1 v) and out_s needed
  for (int k = 0; k < 3; ++k) {
    gemm16(&aV[k][0][0], 32, p.hWv1_f, 32, &fg[0][0], 32, 32, 32);
    __syncthreads();
    FORI(i, 16 * 32) { int c = i % 32, m = i / 32; float v = fg[m][c]; fn[m][c] += v * v; }
    __syncthreads();
  }
  FORI(i, 16 * 32) { int c = i % 32, m = i / 32; hX2[m][c] = (_Float16)sqrtf(fn[m][c]); }
  FORI(i, 16 * 128) { int c = i % 128, m = i / 128; hX2[m][32 + c] = aS[m][c]; }
  __syncthreads();
  FORI(i, 16 * NCLS_) {
    int o = i % NCLS_, m = i / NCLS_;
    float acc = 0.f;
    for (int j = 0; j < 160; ++j) acc += (float)hX2[m][j] * p.cls_gvl_Ws[o * 160 + j];
    p.out[(size_t)(q0 + m) * NCLS_ + o] = acc;
  }
}

// ---------------- host launch ----------------
extern "C" void kernel_launch(void* const* d_in, const int* in_sizes, int n_in,
                              void* d_out, int out_size, void* d_ws, size_t ws_size,
                              hipStream_t stream) {
  (void)in_sizes; (void)n_in; (void)out_size; (void)ws_size;
  P p{};
  p.pos_q      = (const float*)d_in[IN_POSQ];
  p.pos_c      = (const float*)d_in[IN_POSC];
  p.node_sca   = (const float*)d_in[IN_NSCA];
  p.node_vec   = (const float*)d_in[IN_NVEC];
  p.edge_index = (const int*)d_in[IN_EIDX];
  p.W_exp   = (const float*)d_in[P_W_EXP];
  p.b_sca   = (const float*)d_in[P_B_SCA];
  p.b_e2n   = (const float*)d_in[P_B_E2N];
  p.b_n2e   = (const float*)d_in[P_B_N2E];
  p.node_bg = (const float*)d_in[PN_BG];
  p.edge_bg = (const float*)d_in[PE_BG];
  p.out_bg  = (const float*)d_in[PO_BG];
  p.cls_bg  = (const float*)d_in[PCP_BG];
  p.cls_gvl_Ws = (const float*)d_in[PC_GVL_WS];
  p.out = (float*)d_out;

  // workspace carve (256B aligned)
  char* w = (char*)d_ws;
  size_t off = 0;
  auto take = [&](size_t bytes) -> void* {
    off = (off + 255) & ~(size_t)255;
    void* r = w + off; off += bytes; return r;
  };
  _Float16* hWv1_n  = (_Float16*)take(64 * 64 * 2);
  _Float16* hWv2_n  = (_Float16*)take(32 * 64 * 2);
  _Float16* hWs_n   = (_Float16*)take(128 * 320 * 2);
  _Float16* hWg_n   = (_Float16*)take(32 * 128 * 2);
  _Float16* hWdir_e = (_Float16*)take(64 * 64 * 2);
  _Float16* hWg_e   = (_Float16*)take(64 * 64 * 2);
  _Float16* hWs_e   = (_Float16*)take(64 * 128 * 2);
  _Float16* hWv1_e  = (_Float16*)take(64 * 64 * 2);
  _Float16* hWv2_e  = (_Float16*)take(64 * 64 * 2);
  _Float16* hWsca   = (_Float16*)take(128 * 64 * 2);
  _Float16* hWe2n   = (_Float16*)take(32 * 64 * 2);
  _Float16* hWn2e   = (_Float16*)take(32 * 128 * 2);
  _Float16* hWevn   = (_Float16*)take(32 * 64 * 2);
  _Float16* hWg_o   = (_Float16*)take(32 * 128 * 2);
  _Float16* hWs_o   = (_Float16*)take(128 * 160 * 2);
  _Float16* hWv1_o  = (_Float16*)take(32 * 32 * 2);
  _Float16* hWv2_o  = (_Float16*)take(32 * 32 * 2);
  _Float16* hWdir_c = (_Float16*)take(32 * 32 * 2);
  _Float16* hWg_c   = (_Float16*)take(32 * 128 * 2);
  _Float16* hWs_c   = (_Float16*)take(128 * 160 * 2);
  _Float16* hWv1_c  = (_Float16*)take(32 * 32 * 2);
  _Float16* hWv2_c  = (_Float16*)take(32 * 32 * 2);
  _Float16* hWv1_f  = (_Float16*)take(32 * 32 * 2);
  p.nsall = (_Float16*)take((size_t)NCC * 128 * 2);
  p.nvall = (_Float16*)take((size_t)NCC * 96 * 2);
  p.agg_s = (float*)take((size_t)NQC * 128 * 4 + (size_t)NQC * 96 * 4);
  p.agg_v = p.agg_s + (size_t)NQC * 128;

  p.hWv1_n = hWv1_n; p.hWv2_n = hWv2_n; p.hWs_n = hWs_n; p.hWg_n = hWg_n;
  p.hWdir_e = hWdir_e; p.hWg_e = hWg_e; p.hWs_e = hWs_e; p.hWv1_e = hWv1_e; p.hWv2_e = hWv2_e;
  p.hWsca = hWsca; p.hWe2n = hWe2n; p.hWn2e = hWn2e; p.hWevn = hWevn;
  p.hWg_o = hWg_o; p.hWs_o = hWs_o; p.hWv1_o = hWv1_o; p.hWv2_o = hWv2_o;
  p.hWdir_c = hWdir_c; p.hWg_c = hWg_c; p.hWs_c = hWs_c; p.hWv1_c = hWv1_c; p.hWv2_c = hWv2_c;
  p.hWv1_f = hWv1_f;

  auto cvt = [&](int idx, _Float16* dst, int n) {
    cvt_kernel<<<(n + 255) / 256, 256, 0, stream>>>((const float*)d_in[idx], dst, n);
  };
  cvt(PN_WV1, hWv1_n, 64 * 64);    cvt(PN_WV2, hWv2_n, 32 * 64);
  cvt(PN_WS,  hWs_n, 128 * 320);   cvt(PN_WG,  hWg_n, 32 * 128);
  cvt(PE_WDIR, hWdir_e, 64 * 64);  cvt(PE_WG, hWg_e, 64 * 64);
  cvt(PE_WS, hWs_e, 64 * 128);     cvt(PE_WV1, hWv1_e, 64 * 64);
  cvt(PE_WV2, hWv2_e, 64 * 64);
  cvt(P_W_SCA, hWsca, 128 * 64);   cvt(P_W_E2N, hWe2n, 32 * 64);
  cvt(P_W_N2E, hWn2e, 32 * 128);   cvt(P_W_EVN, hWevn, 32 * 64);
  cvt(PO_WG, hWg_o, 32 * 128);     cvt(PO_WS, hWs_o, 128 * 160);
  cvt(PO_WV1, hWv1_o, 32 * 32);    cvt(PO_WV2, hWv2_o, 32 * 32);
  cvt(PCP_WDIR, hWdir_c, 32 * 32); cvt(PCP_WG, hWg_c, 32 * 128);
  cvt(PCP_WS, hWs_c, 128 * 160);   cvt(PCP_WV1, hWv1_c, 32 * 32);
  cvt(PCP_WV2, hWv2_c, 32 * 32);   cvt(PC_GVL_WV1, hWv1_f, 32 * 32);

  {
    int n = NQC * (128 + 96);
    zero_kernel<<<(n + 255) / 256, 256, 0, stream>>>(p.agg_s, n);
  }

  node_kernel<<<NCC / 16, 128, 0, stream>>>(p);
  edge_kernel<<<ETOT / 16, 128, 0, stream>>>(p);
  cls_kernel<<<NQC / 16, 128, 0, stream>>>(p);
}
